// RNNModule_37898791420421
// MI455X (gfx1250) — matrix operand order repaired and checked by reading, not compiled, not run on hardware
//
#include <hip/hip_runtime.h>
#include <hip/hip_bf16.h>

// ---------------------------------------------------------------------------
// Fused 2-layer GRU (PyTorch semantics), B=1024, T=512, I=8, H=50.
// One block = 16 batch rows, 4 waves (wave32). Wave w owns gate-output
// columns [16w, 16w+16) for all three gates -> elementwise update stays in
// the WMMA D layout. h state double-buffered in LDS (f32 authoritative +
// f16 shadow for A operands). Weights staged once into LDS in B-operand
// lane order. v_wmma_f32_16x16x32_f16 with f32 accumulate.
// ---------------------------------------------------------------------------

typedef __attribute__((ext_vector_type(16))) _Float16 v16h;
typedef __attribute__((ext_vector_type(8)))  _Float16 v8h;
typedef __attribute__((ext_vector_type(8)))  float    v8f;

static_assert(sizeof(_Float16) == 2, "f16 size");

#define GB   1024   // batch
#define GT   512    // time steps
#define GI   8      // input size
#define GH   50     // hidden size
#define ROWS 16     // batch rows per block
#define NTH  128    // 4 waves of 32

// staged-weight region bases (1KB blocks of 32 lanes x 16 halfs)
#define RB_X0 0     // layer0 Gx : 3 gates * 4 ntiles * 1 kstep  = 12 blocks
#define RB_H0 12    // layer0 Gh : 3*4*2                          = 24 blocks
#define RB_X1 36    // layer1 Gx : 3*4*2                          = 24 blocks
#define RB_H1 60    // layer1 Gh : 3*4*2                          = 24 blocks
#define NBLK  84

union F8  { v8f  v; float    f[8];  };
union A16 { v16h v; _Float16 h[16]; };

__device__ __forceinline__ v8f wmma_acc(v16h a, v16h b, v8f c) {
  // D = A(16x32 f16) * B(32x16 f16) + C(16x16 f32)
  return __builtin_amdgcn_wmma_f32_16x16x32_f16(false, a, false, b,
                                                (short)0, c, false, false);
}

__device__ __forceinline__ float sigmoid_f(float x) {
  return 1.0f / (1.0f + __expf(-x));
}
__device__ __forceinline__ float tanh_f(float x) {
  float ax = fabsf(x);
  float e  = __expf(-2.0f * ax);
  float r  = (1.0f - e) / (1.0f + e);
  return x < 0.0f ? -r : r;
}

// B operand: 32 lanes x 16 contiguous halfs per lane (32B -> 2x ds_load_b128)
__device__ __forceinline__ v16h load_b(const v16h* sWv, int blk, int lane) {
  return sWv[blk * 32 + lane];
}

// A operand pair (K=64 as two K=32 steps) from a 64-half f16 row in LDS.
// 16-bit A 16x32 layout: lane L (M=L&15), element e -> K = (L>>4)*8 + e  (e<8)
//                                               K = (L>>4)*8 + 16 + (e-8)
__device__ __forceinline__ void load_a_pair(const _Float16* row, int hoff,
                                            v16h& a0, v16h& a1) {
  v8h c0 = *(const v8h*)(row + hoff);
  v8h c1 = *(const v8h*)(row + hoff + 16);
  v8h c2 = *(const v8h*)(row + hoff + 32);
  v8h c3 = *(const v8h*)(row + hoff + 48);
  a0 = __builtin_shufflevector(c0, c1, 0,1,2,3,4,5,6,7,8,9,10,11,12,13,14,15);
  a1 = __builtin_shufflevector(c2, c3, 0,1,2,3,4,5,6,7,8,9,10,11,12,13,14,15);
}

extern __shared__ v16h sWv[];   // 84 * 1KB staged weights (32B aligned base)

__global__ __launch_bounds__(NTH)
void gru2_fused_kernel(const float* __restrict__ x,
                       const float* __restrict__ Wih0, const float* __restrict__ Whh0,
                       const float* __restrict__ bih0, const float* __restrict__ bhh0,
                       const float* __restrict__ Wih1, const float* __restrict__ Whh1,
                       const float* __restrict__ bih1, const float* __restrict__ bhh1,
                       float* __restrict__ out) {
  __shared__ float                 h32s[2][2][ROWS * 64]; // [layer][buf][16x64]
  __shared__ __align__(16) _Float16 h16s[2][2][ROWS * 64];

  _Float16* sW = (_Float16*)sWv;
  const int tid   = threadIdx.x;
  const int lane  = tid & 31;
  const int nt    = tid >> 5;          // wave id == ntile (0..3)
  const int bbase = blockIdx.x * ROWS;

  // ---------------- stage weights into LDS in B-operand order ---------------
  // block layout: [blk][lane][e] ; value = W[g*50 + c][k] with bias at k==Kvalid
  for (int i = tid; i < NBLK * 512; i += NTH) {
    const int blk = i >> 9;
    const int ls  = (i >> 4) & 31;
    const int e   = i & 15;
    const float *W, *bias;
    int kvalid, stride, ksteps, rel;
    if (blk < RB_H0)      { rel = blk;         ksteps = 1; W = Wih0; bias = bih0; kvalid = GI; stride = GI; }
    else if (blk < RB_X1) { rel = blk - RB_H0; ksteps = 2; W = Whh0; bias = bhh0; kvalid = GH; stride = GH; }
    else if (blk < RB_H1) { rel = blk - RB_X1; ksteps = 2; W = Wih1; bias = bih1; kvalid = GH; stride = GH; }
    else                  { rel = blk - RB_H1; ksteps = 2; W = Whh1; bias = bhh1; kvalid = GH; stride = GH; }
    const int g   = rel / (4 * ksteps);
    const int rem = rel % (4 * ksteps);
    const int nts = rem / ksteps;
    const int ks  = rem % ksteps;
    const int c   = nts * 16 + (ls & 15);                 // output column in gate
    const int k   = ks * 32 + ((ls >> 4) << 3) + (e & 7) + ((e >> 3) << 4); // K index (B layout)
    float v = 0.0f;
    if (c < GH) {
      const int R = g * GH + c;
      if (k < kvalid)       v = W[R * stride + k];
      else if (k == kvalid) v = bias[R];
    }
    sW[i] = (_Float16)v;
  }
  // -------- init h state: zeros, f16 shadow has bias 1.0 at col 50 ----------
  for (int i = tid; i < 2 * 2 * ROWS * 64; i += NTH) {
    const int c = i & 63;
    ((float*)h32s)[i]    = 0.0f;
    ((_Float16*)h16s)[i] = (c == GH) ? (_Float16)1.0f : (_Float16)0.0f;
  }
  __syncthreads();

  const int col   = nt * 16 + (lane & 15);   // D-layout column this lane owns
  const int mbase = (lane >> 4) << 3;        // D-layout row base (0 or 8)
  const int hoff  = (lane >> 4) << 3;        // A-layout K base (0 or 8)
  const int mrow  = lane & 15;               // A-layout row (batch row in tile)

  for (int t = 0; t < GT; ++t) {
    const int p = t & 1, q = p ^ 1;

    // ====================== layer 0 ======================
    // A_x: [x_t | 1 | 0...] K=32.  lanes 0-15: e<8 -> x[e]; lanes 16-31: e==0 -> 1.0
    const float* xp = x + (((size_t)(bbase + mrow) * GT + (size_t)t) << 3);
    float4 xa = *(const float4*)xp;
    float4 xb = *(const float4*)(xp + 4);
    if (t + 1 < GT) __builtin_prefetch(xp + 8, 0, 3);
    float xv[8] = {xa.x, xa.y, xa.z, xa.w, xb.x, xb.y, xb.z, xb.w};
    A16 ax;
    const bool hi = lane >= 16;
#pragma unroll
    for (int e = 0; e < 8; ++e)
      ax.h[e] = hi ? ((e == 0) ? (_Float16)1.0f : (_Float16)0.0f) : (_Float16)xv[e];
#pragma unroll
    for (int e = 8; e < 16; ++e) ax.h[e] = (_Float16)0.0f;

    v16h ah0, ah1;
    load_a_pair(&h16s[0][p][mrow * 64], hoff, ah0, ah1);

    {
      F8 hold, gr, gz, hnew;
#pragma unroll
      for (int j = 0; j < 8; ++j) hold.f[j] = h32s[0][p][(mbase + j) * 64 + col];
#pragma unroll
      for (int g = 0; g < 3; ++g) {
        v8f accx = {0.f,0.f,0.f,0.f,0.f,0.f,0.f,0.f};
        accx = wmma_acc(ax.v, load_b(sWv, RB_X0 + (g * 4 + nt), lane), accx);
        v8f acch = {0.f,0.f,0.f,0.f,0.f,0.f,0.f,0.f};
        const int bh = RB_H0 + (g * 4 + nt) * 2;
        acch = wmma_acc(ah0, load_b(sWv, bh + 0, lane), acch);
        acch = wmma_acc(ah1, load_b(sWv, bh + 1, lane), acch);
        F8 X; X.v = accx; F8 Hg; Hg.v = acch;
        if (g == 0) {
#pragma unroll
          for (int j = 0; j < 8; ++j) gr.f[j] = sigmoid_f(X.f[j] + Hg.f[j]);
        } else if (g == 1) {
#pragma unroll
          for (int j = 0; j < 8; ++j) gz.f[j] = sigmoid_f(X.f[j] + Hg.f[j]);
        } else {
#pragma unroll
          for (int j = 0; j < 8; ++j) {
            float n = tanh_f(X.f[j] + gr.f[j] * Hg.f[j]);
            hnew.f[j] = (1.0f - gz.f[j]) * n + gz.f[j] * hold.f[j];
          }
        }
      }
      if (col < GH) {
#pragma unroll
        for (int j = 0; j < 8; ++j) {
          h32s[0][q][(mbase + j) * 64 + col] = hnew.f[j];
          h16s[0][q][(mbase + j) * 64 + col] = (_Float16)hnew.f[j];
        }
      }
    }
    __syncthreads();   // h1(t) visible to all waves

    // ====================== layer 1 ======================
    v16h bx0, bx1;                               // A from fresh h1 (buf q)
    load_a_pair(&h16s[0][q][mrow * 64], hoff, bx0, bx1);
    load_a_pair(&h16s[1][p][mrow * 64], hoff, ah0, ah1);  // A from old h2

    {
      F8 hold, gr, gz, hnew;
#pragma unroll
      for (int j = 0; j < 8; ++j) hold.f[j] = h32s[1][p][(mbase + j) * 64 + col];
#pragma unroll
      for (int g = 0; g < 3; ++g) {
        v8f accx = {0.f,0.f,0.f,0.f,0.f,0.f,0.f,0.f};
        const int bxb = RB_X1 + (g * 4 + nt) * 2;
        accx = wmma_acc(bx0, load_b(sWv, bxb + 0, lane), accx);
        accx = wmma_acc(bx1, load_b(sWv, bxb + 1, lane), accx);
        v8f acch = {0.f,0.f,0.f,0.f,0.f,0.f,0.f,0.f};
        const int bh = RB_H1 + (g * 4 + nt) * 2;
        acch = wmma_acc(ah0, load_b(sWv, bh + 0, lane), acch);
        acch = wmma_acc(ah1, load_b(sWv, bh + 1, lane), acch);
        F8 X; X.v = accx; F8 Hg; Hg.v = acch;
        if (g == 0) {
#pragma unroll
          for (int j = 0; j < 8; ++j) gr.f[j] = sigmoid_f(X.f[j] + Hg.f[j]);
        } else if (g == 1) {
#pragma unroll
          for (int j = 0; j < 8; ++j) gz.f[j] = sigmoid_f(X.f[j] + Hg.f[j]);
        } else {
#pragma unroll
          for (int j = 0; j < 8; ++j) {
            float n = tanh_f(X.f[j] + gr.f[j] * Hg.f[j]);
            hnew.f[j] = (1.0f - gz.f[j]) * n + gz.f[j] * hold.f[j];
          }
        }
      }
      if (col < GH) {
#pragma unroll
        for (int j = 0; j < 8; ++j) {
          h32s[1][q][(mbase + j) * 64 + col] = hnew.f[j];
          h16s[1][q][(mbase + j) * 64 + col] = (_Float16)hnew.f[j];
        }
      }
    }
    __syncthreads();   // h2(t) committed before next step reuses buffers
  }

  // T even -> final h2 lives in buf 0
  for (int i = tid; i < ROWS * GH; i += NTH) {
    const int m = i / GH, c = i % GH;
    out[(size_t)(bbase + m) * GH + c] = h32s[1][0][m * 64 + c];
  }
}

extern "C" void kernel_launch(void* const* d_in, const int* in_sizes, int n_in,
                              void* d_out, int out_size, void* d_ws, size_t ws_size,
                              hipStream_t stream) {
  const float* x    = (const float*)d_in[0];
  const float* Wih0 = (const float*)d_in[1];
  const float* Whh0 = (const float*)d_in[2];
  const float* bih0 = (const float*)d_in[3];
  const float* bhh0 = (const float*)d_in[4];
  const float* Wih1 = (const float*)d_in[5];
  const float* Whh1 = (const float*)d_in[6];
  const float* bih1 = (const float*)d_in[7];
  const float* bhh1 = (const float*)d_in[8];
  float* out = (float*)d_out;

  const size_t shmem = (size_t)NBLK * 512 * sizeof(unsigned short); // 86016 B
  gru2_fused_kernel<<<dim3(GB / ROWS), dim3(NTH), shmem, stream>>>(
      x, Wih0, Whh0, bih0, bhh0, Wih1, Whh1, bih1, bhh1, out);
}
